// DecoderNoTrainer_1726576853149
// MI455X (gfx1250) — matrix operand order repaired and checked by reading, not compiled, not run on hardware
//
#include <hip/hip_runtime.h>
#include <hip/hip_bf16.h>
#include <math.h>

// ---------------- problem constants ----------------
#define BB   256   // batch
#define LL   128   // seq len / steps
#define UU   512   // units
#define EE   256   // embed dim
#define VV   128   // vocab
#define KX   768   // E + U
#define G4U  2048  // 4*U

typedef __bf16 bf16;
typedef __attribute__((ext_vector_type(16))) bf16 v16bf;
typedef __attribute__((ext_vector_type(8)))  bf16 v8bf;
typedef __attribute__((ext_vector_type(8)))  float v8f;

// ---------------- WMMA helpers (CDNA5 16x16x32 bf16, wave32) ----------------
__device__ __forceinline__ v8f wmma_bf16(v16bf a, v16bf b, v8f c) {
  // (neg_a, A, neg_b, B, c_mod, C, reuse_a, reuse_b)
  return __builtin_amdgcn_wmma_f32_16x16x32_bf16(false, a, false, b, (short)0, c, false, false);
}

// A: 16x32 tile, row-major source, row = lane&15, K runs {kb..kb+7, kb+16..kb+23}, kb = (lane>>4)*8
__device__ __forceinline__ v16bf load_a_frag(const bf16* base, int ldk, int k0, int lane) {
  int row = lane & 15;
  const bf16* p = base + (size_t)row * ldk + k0 + ((lane >> 4) << 3);
  v8bf lo = *(const v8bf*)p;
  v8bf hi = *(const v8bf*)(p + 16);
  v16bf a;
#pragma unroll
  for (int i = 0; i < 8; ++i) { a[i] = lo[i]; a[8 + i] = hi[i]; }
  return a;
}

// B: 32x16 tile from W^T stored [N][K]: lane holds column n0+(lane&15),
// K runs {kb..kb+7, kb+16..kb+23}, kb = k0 + (lane>>4)*8  -> two 16B loads
__device__ __forceinline__ v16bf load_b_frag(const bf16* wt, int ldk, int n0, int k0, int lane) {
  const bf16* p = wt + (size_t)(n0 + (lane & 15)) * ldk + k0 + ((lane >> 4) << 3);
  v8bf lo = *(const v8bf*)p;
  v8bf hi = *(const v8bf*)(p + 16);
  v16bf b;
#pragma unroll
  for (int i = 0; i < 8; ++i) { b[i] = lo[i]; b[8 + i] = hi[i]; }
  return b;
}

__device__ __forceinline__ float sigmoidf_(float x) { return 1.0f / (1.0f + __expf(-x)); }

// ---------------- prep kernels ----------------
// Wxh^T[n][k] : n in [0,2048), k in [0,768);  k<256 -> Wx[k][n], else Wh[k-256][n]
__global__ void k_build_wxht(const float* __restrict__ Wx, const float* __restrict__ Wh,
                             bf16* __restrict__ dst) {
  int i = blockIdx.x * blockDim.x + threadIdx.x;
  if (i >= G4U * KX) return;
  int n = i / KX, k = i - n * KX;
  float v = (k < EE) ? Wx[(size_t)k * G4U + n] : Wh[(size_t)(k - EE) * G4U + n];
  dst[i] = (bf16)v;
}

// square transpose to bf16: dst[n*512+k] = src[k*512+n]
__global__ void k_transpose_sq(const float* __restrict__ src, bf16* __restrict__ dst) {
  int i = blockIdx.x * blockDim.x + threadIdx.x;
  if (i >= UU * UU) return;
  int n = i / UU, k = i - n * UU;
  dst[i] = (bf16)src[(size_t)k * UU + n];
}

__global__ void k_cvt_bf16(const float* __restrict__ src, bf16* __restrict__ dst, int n) {
  int i = blockIdx.x * blockDim.x + threadIdx.x;
  if (i < n) dst[i] = (bf16)src[i];
}

__global__ void k_init_state(const float* __restrict__ dec_h, const float* __restrict__ dec_c,
                             const int* __restrict__ sos, bf16* __restrict__ h0,
                             float* __restrict__ c, int* __restrict__ tok) {
  int i = blockIdx.x * blockDim.x + threadIdx.x;
  if (i < BB * UU) { h0[i] = (bf16)dec_h[i]; c[i] = dec_c[i]; }
  if (i < BB) tok[i] = sos[0];
}

// ---------------- one-time: enc_proj = bf16( enc_bf @ W1^T ) ----------------
// A rows = B*L = 32768, K = 512, N = 512.  one 16x16 tile per wave.
__global__ void k_encproj(const bf16* __restrict__ enc_bf, const bf16* __restrict__ w1t,
                          bf16* __restrict__ encp) {
  int wave = blockIdx.x * (blockDim.x >> 5) + (threadIdx.x >> 5);
  int lane = threadIdx.x & 31;
  int mt = wave >> 5;      // 2048 row tiles
  int nt = wave & 31;      // 32 col tiles
  const bf16* abase = enc_bf + (size_t)mt * 16 * UU;
  v8f acc = {};
#pragma unroll 4
  for (int k0 = 0; k0 < UU; k0 += 32) {
    v16bf a = load_a_frag(abase, UU, k0, lane);
    v16bf b = load_b_frag(w1t, UU, nt * 16, k0, lane);
    acc = wmma_bf16(a, b, acc);
  }
  int row0 = mt * 16 + ((lane >> 4) << 3);
  int col  = nt * 16 + (lane & 15);
#pragma unroll
  for (int r = 0; r < 8; ++r)
    encp[(size_t)(row0 + r) * UU + col] = (bf16)acc[r];
}

// ---------------- per-step: fused LSTM GEMM + gates ----------------
// wave computes one 16x16 (batch,unit) tile for all four gates, sharing the A fragment.
// A = [emb[tok] | h_in] (768 K), B = Wxh^T rows {n, n+512, n+1024, n+1536}.
__global__ void k_step_lstm(const bf16* __restrict__ emb_bf, const bf16* __restrict__ wxht,
                            const float* __restrict__ bias, const bf16* __restrict__ h_in,
                            bf16* __restrict__ h_out, float* __restrict__ c,
                            const int* __restrict__ tok) {
  int wave = blockIdx.x * (blockDim.x >> 5) + (threadIdx.x >> 5);
  int lane = threadIdx.x & 31;
  int mt = wave >> 5;      // 16 batch tiles
  int nt = wave & 31;      // 32 unit tiles
  int arow = lane & 15;
  int batch_a = mt * 16 + arow;
  int trow = tok[batch_a];
  const bf16* embrow = emb_bf + (size_t)trow * EE;
  const bf16* hrow   = h_in  + (size_t)batch_a * UU;

  v8f ai = {}, af = {}, ag = {}, ao = {};
  for (int k0 = 0; k0 < KX; k0 += 32) {
    // per-lane A fragment: gather from embedding (k<256) or previous h (k>=256)
    const bf16* p = (k0 < EE) ? (embrow + k0) : (hrow + (k0 - EE));
    p += ((lane >> 4) << 3);
    v8bf lo = *(const v8bf*)p;
    v8bf hi = *(const v8bf*)(p + 16);
    v16bf a;
#pragma unroll
    for (int i = 0; i < 8; ++i) { a[i] = lo[i]; a[8 + i] = hi[i]; }

    v16bf b0 = load_b_frag(wxht, KX, nt * 16,            k0, lane);
    v16bf b1 = load_b_frag(wxht, KX, UU     + nt * 16,   k0, lane);
    v16bf b2 = load_b_frag(wxht, KX, 2 * UU + nt * 16,   k0, lane);
    v16bf b3 = load_b_frag(wxht, KX, 3 * UU + nt * 16,   k0, lane);
    ai = wmma_bf16(a, b0, ai);
    af = wmma_bf16(a, b1, af);
    ag = wmma_bf16(a, b2, ag);
    ao = wmma_bf16(a, b3, ao);
  }

  int col  = nt * 16 + (lane & 15);
  float bi = bias[col], bf_ = bias[UU + col], bg = bias[2 * UU + col], bo = bias[3 * UU + col];
  int row0 = mt * 16 + ((lane >> 4) << 3);
#pragma unroll
  for (int r = 0; r < 8; ++r) {
    size_t idx = (size_t)(row0 + r) * UU + col;
    float ig = sigmoidf_(ai[r] + bi);
    float fg = sigmoidf_(af[r] + bf_);
    float gg = tanhf(ag[r] + bg);
    float og = sigmoidf_(ao[r] + bo);
    float c2 = fg * c[idx] + ig * gg;        // each cell owned by exactly one lane
    c[idx] = c2;
    h_out[idx] = (bf16)(og * tanhf(c2));
  }
}

// ---------------- per-step: hw2 = h2 @ W2^T  (fp32 out) ----------------
__global__ void k_step_hw2(const bf16* __restrict__ h2, const bf16* __restrict__ w2t,
                           float* __restrict__ hw2) {
  int wave = blockIdx.x * (blockDim.x >> 5) + (threadIdx.x >> 5);
  int lane = threadIdx.x & 31;
  int mt = wave >> 5;      // 16 batch tiles
  int nt = wave & 31;      // 32 unit tiles
  const bf16* abase = h2 + (size_t)mt * 16 * UU;
  v8f acc = {};
#pragma unroll 4
  for (int k0 = 0; k0 < UU; k0 += 32) {
    v16bf a = load_a_frag(abase, UU, k0, lane);
    v16bf b = load_b_frag(w2t, UU, nt * 16, k0, lane);
    acc = wmma_bf16(a, b, acc);
  }
  int row0 = mt * 16 + ((lane >> 4) << 3);
  int col  = nt * 16 + (lane & 15);
#pragma unroll
  for (int r = 0; r < 8; ++r)
    hw2[(size_t)(row0 + r) * UU + col] = acc[r];
}

// ---------------- per-step: scores -> softmax -> argmax feedback ----------------
// one block per batch row (128 threads: thread = position l)
__global__ void k_step_attn(const bf16* __restrict__ encp, const float* __restrict__ hw2,
                            const float* __restrict__ v, float* __restrict__ out,
                            int* __restrict__ tok, int t) {
  __shared__ float sv[LL];
  __shared__ int   si[LL];
  int b = blockIdx.x;
  int l = threadIdx.x;
  const bf16*  ep = encp + ((size_t)b * LL + l) * UU;
  const float* hw = hw2 + (size_t)b * UU;   // uniform across block -> scalar path
  float s = 0.0f;
#pragma unroll 4
  for (int k = 0; k < UU; ++k)
    s += v[k] * tanhf((float)ep[k] + hw[k]);

  // max + argmax (tie -> lowest index, matching jnp.argmax)
  sv[l] = s; si[l] = l;
  __syncthreads();
  for (int off = LL / 2; off > 0; off >>= 1) {
    if (l < off) {
      float o = sv[l + off]; int oi = si[l + off];
      if (o > sv[l] || (o == sv[l] && oi < si[l])) { sv[l] = o; si[l] = oi; }
    }
    __syncthreads();
  }
  float mx = sv[0]; int am = si[0];
  __syncthreads();
  float e = __expf(s - mx);
  sv[l] = e;
  __syncthreads();
  for (int off = LL / 2; off > 0; off >>= 1) {
    if (l < off) sv[l] += sv[l + off];
    __syncthreads();
  }
  out[((size_t)b * LL + t) * LL + l] = e / sv[0];
  if (l == 0) tok[b] = am;
}

// ---------------- host launcher ----------------
extern "C" void kernel_launch(void* const* d_in, const int* in_sizes, int n_in,
                              void* d_out, int out_size, void* d_ws, size_t ws_size,
                              hipStream_t stream) {
  const float* dec_h = (const float*)d_in[0];
  const float* dec_c = (const float*)d_in[1];
  const float* enc_o = (const float*)d_in[2];
  // d_in[3] = enc_input (unused by the reference)
  const float* emb   = (const float*)d_in[4];
  const float* Wx    = (const float*)d_in[5];
  const float* Wh    = (const float*)d_in[6];
  const float* bias  = (const float*)d_in[7];
  const float* W1    = (const float*)d_in[8];
  const float* W2    = (const float*)d_in[9];
  const float* vvec  = (const float*)d_in[10];
  const int*   sos   = (const int*)d_in[11];
  float* out = (float*)d_out;

  // workspace layout (all offsets 256B aligned)
  char* w = (char*)d_ws;
  bf16* wxht   = (bf16*)(w + 0);                       // 2048*768*2  = 3,145,728
  bf16* w2t    = (bf16*)(w + 3145728);                 // 512*512*2   =   524,288
  bf16* w1t    = (bf16*)(w + 3670016);                 // 512*512*2   =   524,288
  bf16* emb_bf = (bf16*)(w + 4194304);                 // 128*256*2   =    65,536
  bf16* hb0    = (bf16*)(w + 4259840);                 // 256*512*2   =   262,144
  bf16* hb1    = (bf16*)(w + 4521984);                 // 256*512*2   =   262,144
  float* cbuf  = (float*)(w + 4784128);                // 256*512*4   =   524,288
  float* hw2   = (float*)(w + 5308416);                // 256*512*4   =   524,288
  int*  tok    = (int*)(w + 5832704);                  //                  1,024
  bf16* enc_bf = (bf16*)(w + 5833728);                 // 32768*512*2 = 33,554,432
  bf16* encp   = (bf16*)(w + 39388160);                // 32768*512*2 = 33,554,432
  (void)in_sizes; (void)n_in; (void)out_size; (void)ws_size;

  // ---- one-time prep ----
  k_build_wxht <<<(G4U * KX + 255) / 256, 256, 0, stream>>>(Wx, Wh, wxht);
  k_transpose_sq<<<(UU * UU + 255) / 256, 256, 0, stream>>>(W2, w2t);
  k_transpose_sq<<<(UU * UU + 255) / 256, 256, 0, stream>>>(W1, w1t);
  k_cvt_bf16   <<<(VV * EE + 255) / 256, 256, 0, stream>>>(emb, emb_bf, VV * EE);
  k_cvt_bf16   <<<(BB * LL * UU + 255) / 256, 256, 0, stream>>>(enc_o, enc_bf, BB * LL * UU);
  k_init_state <<<(BB * UU + 255) / 256, 256, 0, stream>>>(dec_h, dec_c, sos, hb0, cbuf, tok);

  // enc_proj: 2048 x 32 tiles, 8 waves per 256-thread block -> 8192 blocks
  k_encproj<<<8192, 256, 0, stream>>>(enc_bf, w1t, encp);

  // ---- 128 sequential decode steps ----
  for (int t = 0; t < LL; ++t) {
    bf16* h_in  = (t & 1) ? hb1 : hb0;
    bf16* h_out = (t & 1) ? hb0 : hb1;
    // 16x32 tiles, 8 waves/block -> 64 blocks
    k_step_lstm<<<64, 256, 0, stream>>>(emb_bf, wxht, bias, h_in, h_out, cbuf, tok);
    k_step_hw2 <<<64, 256, 0, stream>>>(h_out, w2t, hw2);
    k_step_attn<<<BB, LL, 0, stream>>>(encp, hw2, vvec, out, tok, t);
  }
}
